// ConfigurableSALayer_11596411699406
// MI455X (gfx1250) — matrix operand order repaired and checked by reading, not compiled, not run on hardware
//
#include <hip/hip_runtime.h>
#include <math.h>

// ---------------- problem constants ----------------
#define BB       8
#define NN       4096
#define C_IN     64
#define NPOINT   1024
#define KNN      32
#define NCOLS    (BB * NPOINT * KNN)     // 262144 GEMM columns
#define NS       (BB * NPOINT)           // 8192 (b,s) pairs
#define C0       64
#define C1       64
#define C2       128
#define KP_IN    96                      // 67 padded to mult of 32
#define BN_INV   0.99999499997f          // 1/sqrt(1+1e-5)

typedef _Float16 f16;
typedef __attribute__((ext_vector_type(16))) _Float16 v16h;
typedef __attribute__((ext_vector_type(8)))  _Float16 v8h;
typedef __attribute__((ext_vector_type(8)))  float    v8f;

__device__ __forceinline__ float sigmoidf_(float x) { return 1.0f / (1.0f + expf(-x)); }

// ============================================================
// WMMA GEMM:  Out[n][m] = epilogue( sum_k W[m][k] * X[n][k] )
//  W: f16 [M][KP]  (A matrix, row-major, K contiguous)
//  X: f16 [N][KP]  (B matrix, K contiguous per column)
//  MODE 0: raw f32 store          (used for U = (W1-W2) @ center)
//  MODE 1: bn+relu -> f16 store   (conv0/1/2, sp1)
//  MODE 2: +U[bs][m], bn+relu -> f16 (edge conv)
//  MTW:   M-tiles accumulated per wave; one B fragment is reused
//         across MTW WMMAs (register blocking on M).
//  Compile-time KP/M: K-loop fully unrolls; per K-step all loads
//  (B + MTW A fragments) are issued as one clause, then MTW WMMAs
//  run back-to-back, allowing partial loadcnt waits.
// ============================================================
template <int MODE, int MTW, int KP, int M>
__global__ void wmma_gemm(const f16* __restrict__ W, const f16* __restrict__ X,
                          void* __restrict__ Out,
                          const float* __restrict__ rowScale,
                          const float* __restrict__ rowBias,
                          const float* __restrict__ U)
{
    const int lane = threadIdx.x & 31;
    const int wave = blockIdx.x * (blockDim.x >> 5) + (threadIdx.x >> 5);
    constexpr int mGroups = M / (16 * MTW);
    const int mG = wave % mGroups;
    const int nT = wave / mGroups;
    const int m0 = mG * 16 * MTW;
    const int n0 = nT << 4;
    const int half = lane >> 4;      // 0: lanes 0-15, 1: lanes 16-31
    const int lm   = lane & 15;

    // A: lane row = m+lm; holds K = half*16..+15 of each 32-chunk (contiguous)
    const f16* wBase = W + (size_t)(m0 + lm) * KP + half * 16;
    // B: lane col = n0+lm; holds K = half*16..+15 of each 32-chunk (contiguous)
    const f16* xRow  = X + (size_t)(n0 + lm) * KP + half * 16;

    v8f acc[MTW];
    #pragma unroll
    for (int t = 0; t < MTW; ++t) acc[t] = (v8f){};

    union AFrag { v16h v; v8h h[2]; };

    #pragma unroll
    for (int k0 = 0; k0 < KP; k0 += 32) {
        // ---- load clause: B fragment + all MTW A fragments ----
        v16h b = *(const v16h*)(xRow + k0);
        __builtin_prefetch(xRow + k0 + 32, 0, 1);
        AFrag a[MTW];
        #pragma unroll
        for (int t = 0; t < MTW; ++t) {
            const f16* wr = wBase + (size_t)(t * 16) * KP + k0;
            a[t].v = *(const v16h*)(wr);
        }
        // ---- matrix clause: MTW back-to-back WMMAs ----
        #pragma unroll
        for (int t = 0; t < MTW; ++t) {
            acc[t] = __builtin_amdgcn_wmma_f32_16x16x32_f16(
                false, a[t].v, false, b, (short)0, acc[t], false, false);
        }
    }

    const int n = n0 + lm;
    #pragma unroll
    for (int t = 0; t < MTW; ++t) {
        const int mb = m0 + t * 16 + half * 8;   // C/D layout: vgpr r -> M = r (+8 upper)
        if (MODE == 0) {
            float* o = (float*)Out + (size_t)n * M + mb;
            #pragma unroll
            for (int r = 0; r < 8; ++r) o[r] = acc[t][r];
        } else {
            v8h ov;
            #pragma unroll
            for (int r = 0; r < 8; ++r) {
                float v = acc[t][r];
                if (MODE == 2) v += U[(size_t)(n >> 5) * M + (mb + r)];
                v = v * rowScale[mb + r] + rowBias[mb + r];
                v = fmaxf(v, 0.0f);
                ov[r] = (f16)v;
            }
            *(v8h*)((f16*)Out + (size_t)n * M + mb) = ov;
        }
    }
}

// ============================================================
// Farthest point sampling: 1 block (1024 thr) per batch.
// dist kept in registers (4 pts/thread), LDS argmax tree per step.
// Also emits new_xyz (ws, [bs][3]) and d_out new_xyz^T (B,3,S).
// ============================================================
__global__ void fps_kernel(const float* __restrict__ xyz, int* __restrict__ cidx,
                           float* __restrict__ newxyz, float* __restrict__ out0)
{
    __shared__ float sval[1024];
    __shared__ int   sidx[1024];
    __shared__ float cpt[3];
    __shared__ int   sfar;
    const int b = blockIdx.x, t = threadIdx.x;
    const float* px = xyz + (size_t)b * 3 * NN;
    const float* py = px + NN;
    const float* pz = py + NN;
    float dist[4];
    #pragma unroll
    for (int j = 0; j < 4; ++j) dist[j] = 1e10f;
    if (t == 0) sfar = 0;
    __syncthreads();

    for (int it = 0; it < NPOINT; ++it) {
        const int far = sfar;
        if (t == 0) {
            float cx = px[far], cy = py[far], cz = pz[far];
            cidx[b * NPOINT + it] = far;
            cpt[0] = cx; cpt[1] = cy; cpt[2] = cz;
            float* nz = newxyz + (size_t)(b * NPOINT + it) * 3;
            nz[0] = cx; nz[1] = cy; nz[2] = cz;
            float* o = out0 + (size_t)b * 3 * NPOINT;
            o[0 * NPOINT + it] = cx; o[1 * NPOINT + it] = cy; o[2 * NPOINT + it] = cz;
        }
        __syncthreads();
        const float cx = cpt[0], cy = cpt[1], cz = cpt[2];
        float best = -1.0f; int bi = 0;
        #pragma unroll
        for (int j = 0; j < 4; ++j) {
            const int n = t + j * 1024;
            float dx = px[n] - cx, dy = py[n] - cy, dz = pz[n] - cz;
            float d = dx * dx + dy * dy + dz * dz;
            dist[j] = fminf(dist[j], d);
            if (dist[j] > best) { best = dist[j]; bi = n; }
        }
        sval[t] = best; sidx[t] = bi;
        __syncthreads();
        for (int s = 512; s > 0; s >>= 1) {
            if (t < s) {
                float v2 = sval[t + s]; int i2 = sidx[t + s];
                if (v2 > sval[t] || (v2 == sval[t] && i2 < sidx[t])) {
                    sval[t] = v2; sidx[t] = i2;
                }
            }
            __syncthreads();
        }
        if (t == 0) sfar = sidx[0];
        __syncthreads();
    }
}

// ---------------- KNN: 1 thread per query, top-32 insertion ----------------
__global__ void knn_kernel(const float* __restrict__ xyz,
                           const float* __restrict__ newxyz, int* __restrict__ gidx)
{
    const int q = blockIdx.x * blockDim.x + threadIdx.x;   // 0..8191
    const int b = q >> 10;
    const float* px = xyz + (size_t)b * 3 * NN;
    const float* py = px + NN;
    const float* pz = py + NN;
    const float qx = newxyz[q * 3 + 0], qy = newxyz[q * 3 + 1], qz = newxyz[q * 3 + 2];
    float nd[KNN]; int ni[KNN];
    #pragma unroll
    for (int k = 0; k < KNN; ++k) { nd[k] = 3e38f; ni[k] = 0; }
    for (int n = 0; n < NN; ++n) {
        float dx = px[n] - qx, dy = py[n] - qy, dz = pz[n] - qz;
        float d = dx * dx + dy * dy + dz * dz;
        if (d < nd[KNN - 1]) {
            int j = KNN - 1;
            while (j > 0 && nd[j - 1] > d) { nd[j] = nd[j - 1]; ni[j] = ni[j - 1]; --j; }
            nd[j] = d; ni[j] = n;
        }
    }
    int* o = gidx + (size_t)q * KNN;
    #pragma unroll
    for (int k = 0; k < KNN; ++k) o[k] = ni[k];
}

// ---------------- gather + concat -> XT f16 [NCOLS][96] ----------------
__global__ void gather_kernel(const float* __restrict__ xyz, const float* __restrict__ points,
                              const float* __restrict__ newxyz, const int* __restrict__ gidx,
                              f16* __restrict__ XT)
{
    const int n = blockIdx.x * blockDim.x + threadIdx.x;    // 0..NCOLS-1
    const int b = n >> 15;
    const int s = (n >> 5) & 1023;
    const int idx = gidx[n];
    const float* px = xyz + (size_t)b * 3 * NN;
    const float* nz = newxyz + (size_t)(b * NPOINT + s) * 3;
    f16* o = XT + (size_t)n * KP_IN;
    o[0] = (f16)(px[idx] - nz[0]);
    o[1] = (f16)(px[NN + idx] - nz[1]);
    o[2] = (f16)(px[2 * NN + idx] - nz[2]);
    const float* pp = points + (size_t)b * C_IN * NN + idx;
    #pragma unroll 4
    for (int c = 0; c < C_IN; ++c) o[3 + c] = (f16)pp[(size_t)c * NN];
    #pragma unroll
    for (int c = 3 + C_IN; c < KP_IN; ++c) o[c] = (f16)0.0f;
}

// ---------------- weight conversion / prep ----------------
__global__ void cvt_w(const float* __restrict__ src, f16* __restrict__ dst,
                      int M, int K, int Kpad)
{
    const int i = blockIdx.x * blockDim.x + threadIdx.x;
    if (i >= M * Kpad) return;
    const int m = i / Kpad, k = i - m * Kpad;
    dst[i] = (k < K) ? (f16)src[m * K + k] : (f16)0.0f;
}

__global__ void edge_split(const float* __restrict__ edge_w,
                           f16* __restrict__ WE, f16* __restrict__ WD)
{
    const int i = blockIdx.x * blockDim.x + threadIdx.x;     // 0..16383
    const int m = i >> 7, c = i & 127;
    const float w1 = edge_w[m * 256 + c];
    const float w2 = edge_w[m * 256 + 128 + c];
    WE[i] = (f16)w2;
    WD[i] = (f16)(w1 - w2);
}

__global__ void fold_bn(const float* __restrict__ g, const float* __restrict__ bnb,
                        const float* __restrict__ cb, float* __restrict__ sc,
                        float* __restrict__ bi, int M)
{
    const int m = blockIdx.x * blockDim.x + threadIdx.x;
    if (m >= M) return;
    const float s = g[m] * BN_INV;
    sc[m] = s;
    bi[m] = (cb ? cb[m] * s : 0.0f) + bnb[m];
}

// ---------------- SE layer ----------------
__global__ void se_max(const f16* __restrict__ Y0, float* __restrict__ y0max)
{
    __shared__ float red[256];
    const int u = blockIdx.x;           // 0..511 : (b,c)
    const int b = u >> 6, c = u & 63;
    const int t = threadIdx.x;
    const f16* base = Y0 + (size_t)b * (NPOINT * KNN) * C0 + c;
    float mx = -3e38f;
    for (int i = t; i < NPOINT * KNN; i += 256)
        mx = fmaxf(mx, (float)base[(size_t)i * C0]);
    red[t] = mx;
    __syncthreads();
    for (int s = 128; s > 0; s >>= 1) {
        if (t < s) red[t] = fmaxf(red[t], red[t + s]);
        __syncthreads();
    }
    if (t == 0) y0max[u] = red[0];
}

__global__ void se_mlp(const float* __restrict__ y0max, const float* __restrict__ w1,
                       const float* __restrict__ w2, float* __restrict__ sescale)
{
    __shared__ float h[4];
    const int b = blockIdx.x, t = threadIdx.x;   // 64 threads
    if (t < 4) {
        float a = 0.0f;
        for (int c = 0; c < 64; ++c) a += w1[t * 64 + c] * y0max[b * 64 + c];
        h[t] = fmaxf(a, 0.0f);
    }
    __syncthreads();
    float o = 0.0f;
    #pragma unroll
    for (int r = 0; r < 4; ++r) o += w2[t * 4 + r] * h[r];
    sescale[b * 64 + t] = sigmoidf_(o);
}

__global__ void se_apply(f16* __restrict__ Y0, const float* __restrict__ sescale)
{
    const size_t i = (size_t)blockIdx.x * blockDim.x + threadIdx.x;   // NCOLS*64
    const int c = (int)(i & 63);
    const int n = (int)(i >> 6);
    const int b = n >> 15;
    Y0[i] = (f16)((float)Y0[i] * sescale[b * 64 + c]);
}

// ---------------- CBAM helpers ----------------
__global__ void center_extract(const f16* __restrict__ Y2, f16* __restrict__ CTR)
{
    const int i = blockIdx.x * blockDim.x + threadIdx.x;   // NS*128
    const int bs = i >> 7, c = i & 127;
    CTR[i] = Y2[(size_t)bs * KNN * C2 + c];
}

__global__ void ms_sigmoid(const f16* __restrict__ MS, const float* __restrict__ sp2w,
                           const float* __restrict__ sp2b, float* __restrict__ MsSig)
{
    const int n = blockIdx.x * blockDim.x + threadIdx.x;   // NCOLS
    float acc = sp2b[0];
    const f16* r = MS + (size_t)n * 32;
    #pragma unroll 8
    for (int c = 0; c < 32; ++c) acc += sp2w[c] * (float)r[c];
    MsSig[n] = sigmoidf_(acc);
}

__global__ void fsp_kernel(const f16* __restrict__ Y2, const float* __restrict__ MsSig,
                           f16* __restrict__ FSP)
{
    const size_t i = (size_t)blockIdx.x * blockDim.x + threadIdx.x;   // NCOLS*128
    const int n = (int)(i >> 7);
    FSP[i] = (f16)((float)Y2[i] * MsSig[n]);
}

__global__ void avgmax_kernel(const f16* __restrict__ FSP, float* __restrict__ AVG,
                              float* __restrict__ MX)
{
    const int i = blockIdx.x * blockDim.x + threadIdx.x;   // NS*128
    const int bs = i >> 7, c = i & 127;
    const f16* base = FSP + (size_t)bs * KNN * C2 + c;
    float sum = 0.0f, mx = -3e38f;
    #pragma unroll 8
    for (int k = 0; k < KNN; ++k) {
        float v = (float)base[(size_t)k * C2];
        sum += v; mx = fmaxf(mx, v);
    }
    AVG[i] = sum * (1.0f / KNN);
    MX[i] = mx;
}

__global__ void mc_kernel(const float* __restrict__ AVG, const float* __restrict__ MX,
                          const float* __restrict__ chw1, const float* __restrict__ chw2,
                          float* __restrict__ MC)
{
    const int q = blockIdx.x * blockDim.x + threadIdx.x;   // NS
    const float* av = AVG + (size_t)q * 128;
    const float* mv = MX + (size_t)q * 128;
    float h[8];
    #pragma unroll
    for (int r = 0; r < 8; ++r) {
        float a1 = 0.0f, a2 = 0.0f;
        for (int c = 0; c < 128; ++c) {
            const float w = chw1[r * 128 + c];
            a1 += w * av[c]; a2 += w * mv[c];
        }
        h[r] = fmaxf(a1, 0.0f) + fmaxf(a2, 0.0f);
    }
    float* o = MC + (size_t)q * 128;
    for (int c = 0; c < 128; ++c) {
        float acc = 0.0f;
        #pragma unroll
        for (int r = 0; r < 8; ++r) acc += chw2[c * 8 + r] * h[r];
        o[c] = sigmoidf_(acc);
    }
}

__global__ void agg_kernel(const f16* __restrict__ FSP, const f16* __restrict__ Y2,
                           const float* __restrict__ MC, float* __restrict__ out)
{
    const int i = blockIdx.x * blockDim.x + threadIdx.x;   // NS*128
    const int bs = i >> 7, c = i & 127;
    const int b = bs >> 10, s = bs & 1023;
    const float mc = MC[i];
    const f16* fb = FSP + (size_t)bs * KNN * C2 + c;
    const f16* lb = Y2 + (size_t)bs * KNN * C2 + c;
    float mx = -3e38f;
    #pragma unroll 8
    for (int k = 0; k < KNN; ++k) {
        const size_t off = (size_t)k * C2;
        float f = (float)fb[off] * mc + (float)lb[off];
        mx = fmaxf(mx, f);
    }
    out[(size_t)BB * 3 * NPOINT + ((size_t)b * C2 + c) * NPOINT + s] = mx;
}

// ============================================================
// host-side launch
// ============================================================
extern "C" void kernel_launch(void* const* d_in, const int* in_sizes, int n_in,
                              void* d_out, int out_size, void* d_ws, size_t ws_size,
                              hipStream_t stream)
{
    const float* xyz    = (const float*)d_in[0];
    const float* points = (const float*)d_in[1];
    const float* c0w = (const float*)d_in[2],  *c0b = (const float*)d_in[3];
    const float* g0  = (const float*)d_in[4],  *b0  = (const float*)d_in[5];
    const float* c1w = (const float*)d_in[6],  *c1b = (const float*)d_in[7];
    const float* g1  = (const float*)d_in[8],  *b1  = (const float*)d_in[9];
    const float* c2w = (const float*)d_in[10], *c2b = (const float*)d_in[11];
    const float* g2  = (const float*)d_in[12], *b2  = (const float*)d_in[13];
    const float* sew1 = (const float*)d_in[14], *sew2 = (const float*)d_in[15];
    const float* ew  = (const float*)d_in[16];
    const float* eg  = (const float*)d_in[17], *eb = (const float*)d_in[18];
    const float* s1w = (const float*)d_in[19];
    const float* s1g = (const float*)d_in[20], *s1b = (const float*)d_in[21];
    const float* s2w = (const float*)d_in[22], *s2b = (const float*)d_in[23];
    const float* chw1 = (const float*)d_in[24], *chw2 = (const float*)d_in[25];
    float* out = (float*)d_out;
    char* ws = (char*)d_ws;

    // ---- workspace layout (bytes) ----
    int*   CIDX   = (int*)(ws + 0);                      //  32 KB
    float* NEWXYZ = (float*)(ws + 32768);                //  96 KB
    int*   GIDX   = (int*)(ws + 131072);                 //   1 MB
    f16*   W0h = (f16*)(ws + 1179648);                   //  64x96
    f16*   W1h = (f16*)(ws + 1191936);                   //  64x64
    f16*   W2h = (f16*)(ws + 1200128);                   // 128x64
    f16*   WEh = (f16*)(ws + 1216512);                   // 128x128
    f16*   WDh = (f16*)(ws + 1249280);                   // 128x128
    f16*   WSh = (f16*)(ws + 1282048);                   //  32x128
    float* PAR = (float*)(ws + 1290240);                 // folded bn params
    float* SC0 = PAR, *BI0 = PAR + 64;
    float* SC1 = PAR + 128, *BI1 = PAR + 192;
    float* SC2 = PAR + 256, *BI2 = PAR + 384;
    float* SCE = PAR + 512, *BIE = PAR + 640;
    float* SCS = PAR + 768, *BIS = PAR + 800;
    float* Y0MAX   = (float*)(ws + 1310720);
    float* SESCALE = (float*)(ws + 1314816);
    f16*   XT  = (f16*)(ws + 2097152);                   //  48 MB
    f16*   Y0  = (f16*)(ws + 52428800ull);               //  32 MB
    f16*   Y1  = (f16*)(ws + 85983232ull);               //  32 MB
    f16*   Y2  = (f16*)(ws + 119537664ull);              //  64 MB
    f16*   CTR = (f16*)(ws + 186646528ull);              //   2 MB
    float* U   = (float*)(ws + 188743680ull);            //   4 MB
    f16*   E   = (f16*)(ws + 192937984ull);              //  64 MB
    f16*   MS  = (f16*)(ws + 260046848ull);              //  16 MB
    float* MSG = (float*)(ws + 276824064ull);            //   1 MB
    f16*   FSP = (f16*)(ws + 277872640ull);              //  64 MB
    float* AVG = (float*)(ws + 344981504ull);            //   4 MB
    float* MX  = (float*)(ws + 349175808ull);            //   4 MB
    float* MC  = (float*)(ws + 353370112ull);            //   4 MB

    // ---- weight prep ----
    cvt_w<<<(64 * 96 + 255) / 256, 256, 0, stream>>>(c0w, W0h, 64, 67, 96);
    cvt_w<<<(64 * 64 + 255) / 256, 256, 0, stream>>>(c1w, W1h, 64, 64, 64);
    cvt_w<<<(128 * 64 + 255) / 256, 256, 0, stream>>>(c2w, W2h, 128, 64, 64);
    cvt_w<<<(32 * 128 + 255) / 256, 256, 0, stream>>>(s1w, WSh, 32, 128, 128);
    edge_split<<<64, 256, 0, stream>>>(ew, WEh, WDh);
    fold_bn<<<1, 64, 0, stream>>>(g0, b0, c0b, SC0, BI0, 64);
    fold_bn<<<1, 64, 0, stream>>>(g1, b1, c1b, SC1, BI1, 64);
    fold_bn<<<1, 128, 0, stream>>>(g2, b2, c2b, SC2, BI2, 128);
    fold_bn<<<1, 128, 0, stream>>>(eg, eb, nullptr, SCE, BIE, 128);
    fold_bn<<<1, 32, 0, stream>>>(s1g, s1b, nullptr, SCS, BIS, 32);

    // ---- sampling / grouping ----
    fps_kernel<<<BB, 1024, 0, stream>>>(xyz, CIDX, NEWXYZ, out);
    knn_kernel<<<NS / 256, 256, 0, stream>>>(xyz, NEWXYZ, GIDX);
    gather_kernel<<<NCOLS / 256, 256, 0, stream>>>(xyz, points, NEWXYZ, GIDX, XT);

    // ---- conv0 + bn + relu  (M=64, K=96): 1 wave -> 64x16 strip ----
    wmma_gemm<1, 4, 96, 64><<<(NCOLS / 16) / 8, 256, 0, stream>>>(
        W0h, XT, Y0, SC0, BI0, nullptr);

    // ---- SE layer ----
    se_max<<<512, 256, 0, stream>>>(Y0, Y0MAX);
    se_mlp<<<8, 64, 0, stream>>>(Y0MAX, sew1, sew2, SESCALE);
    se_apply<<<(NCOLS * 64) / 256, 256, 0, stream>>>(Y0, SESCALE);

    // ---- conv1 (M=64), conv2 (M=128) ----
    wmma_gemm<1, 4, 64, 64><<<(NCOLS / 16) / 8, 256, 0, stream>>>(
        W1h, Y0, Y1, SC1, BI1, nullptr);
    wmma_gemm<1, 4, 64, 128><<<(NCOLS / 16) * 2 / 8, 256, 0, stream>>>(
        W2h, Y1, Y2, SC2, BI2, nullptr);

    // ---- CBAM: edge conv split into U=(W1-W2)@center  +  W2@F ----
    center_extract<<<(NS * 128) / 256, 256, 0, stream>>>(Y2, CTR);
    wmma_gemm<0, 4, 128, 128><<<(NS / 16) * 2 / 8, 256, 0, stream>>>(
        WDh, CTR, U, nullptr, nullptr, nullptr);
    wmma_gemm<2, 4, 128, 128><<<(NCOLS / 16) * 2 / 8, 256, 0, stream>>>(
        WEh, Y2, E, SCE, BIE, U);

    // ---- spatial attention (M=32) ----
    wmma_gemm<1, 2, 128, 32><<<(NCOLS / 16) / 8, 256, 0, stream>>>(
        WSh, E, MS, SCS, BIS, nullptr);
    ms_sigmoid<<<NCOLS / 256, 256, 0, stream>>>(MS, s2w, s2b, MSG);
    fsp_kernel<<<(NCOLS * 128) / 256, 256, 0, stream>>>(Y2, MSG, FSP);

    // ---- channel attention + final aggregation ----
    avgmax_kernel<<<(NS * 128) / 256, 256, 0, stream>>>(FSP, AVG, MX);
    mc_kernel<<<NS / 256, 256, 0, stream>>>(AVG, MX, chw1, chw2, MC);
    agg_kernel<<<(NS * 128) / 256, 256, 0, stream>>>(FSP, Y2, MC, out);
}
